// CrossAttention_45397804319205
// MI455X (gfx1250) — compile-verified, matrix operand order checked by reading, not run on hardware
//
#include <hip/hip_runtime.h>
#include <hip/hip_bf16.h>

// ---------------------------------------------------------------------------
// CDNA5 (gfx1250) cross-attention forward, bf16 WMMA everywhere.
//  - Projection / output GEMMs: 128x128 block tile, 32x64 wave tile,
//    double-buffered LDS with register staging (1 barrier / K-step).
//  - Attention: flash style, K chunk staged via global_load_async_to_lds_b32.
// ---------------------------------------------------------------------------

typedef __attribute__((ext_vector_type(16))) __bf16       v16bf;
typedef __attribute__((ext_vector_type(8)))  float        v8f;
typedef __attribute__((ext_vector_type(8)))  unsigned int v8u;

union FragU { v8u u; v16bf b; };

__device__ __forceinline__ unsigned short f2bf(float f) {
    unsigned int u = __float_as_uint(f);
    u += 0x7fffu + ((u >> 16) & 1u);   // round to nearest even
    return (unsigned short)(u >> 16);
}
__device__ __forceinline__ unsigned int pack2(float a, float b) {
    return (unsigned int)f2bf(a) | ((unsigned int)f2bf(b) << 16);
}

// Async global->LDS copy (CDNA5): per-lane 32-bit, tracked by ASYNCcnt.
__device__ __forceinline__ void async_load_b32(unsigned lds_byte_off,
                                               unsigned long long base,
                                               unsigned byte_off) {
    asm volatile("global_load_async_to_lds_b32 %0, %1, %2"
                 :: "v"(lds_byte_off), "v"(byte_off), "s"(base)
                 : "memory");
}
__device__ __forceinline__ void wait_async() {
#if __has_builtin(__builtin_amdgcn_s_wait_asynccnt)
    __builtin_amdgcn_s_wait_asynccnt(0);
#else
    asm volatile("s_wait_asynccnt 0x0" ::: "memory");
#endif
}

// ---------------------------------------------------------------------------
// Generic GEMM: D[M,N] = A[M,K] @ W[N,K]^T + bias[N]
//   A_BF16  : A is bf16 (else fp32, converted on the fly)
//   SCATTER : write bf16 into [b, h, t, d] head layout (params T,H,HD)
//             else write fp32 contiguous [M, N]
// Block: 256 threads (8 waves), tile 128(M) x 128(N), BK = 32.
// Waves: 4 in M (32 rows each) x 2 in N (64 cols each) -> 8 WMMA / K-step.
// ---------------------------------------------------------------------------
template <bool A_BF16, bool SCATTER>
__global__ __launch_bounds__(256)
void gemm_wmma(const void* __restrict__ Aptr, const float* __restrict__ W,
               const float* __restrict__ bias, void* __restrict__ Dptr,
               int M, int N, int K, int T, int H, int HD)
{
    constexpr int BM = 128, BN = 128, BK = 32;
    constexpr int LDP = BK / 2 + 1;                   // 17 u32, padded
    __shared__ unsigned int As[2][BM][LDP];
    __shared__ unsigned int Ws[2][BN][LDP];

    const int tid  = threadIdx.x;
    const int wave = tid >> 5;
    const int lane = tid & 31;
    const int l16  = lane & 15;
    const int half = lane >> 4;
    const int waveM = wave & 3;                       // 0..3 -> 32-row slice
    const int waveN = wave >> 2;                      // 0..1 -> 64-col slice
    const int rowBase = blockIdx.x * BM;
    const int colBase = blockIdx.y * BN;

    const float*        Af = (const float*)Aptr;
    const unsigned int* Ab = (const unsigned int*)Aptr;  // bf16 pair view

    v8f acc[2][4];
#pragma unroll
    for (int mi = 0; mi < 2; ++mi)
#pragma unroll
        for (int j = 0; j < 4; ++j)
#pragma unroll
            for (int e = 0; e < 8; ++e) acc[mi][j][e] = 0.0f;

    unsigned int aR[8], wR[8];

    auto load_tiles = [&](int k0) {
#pragma unroll
        for (int i = 0; i < 8; ++i) {                 // A: 128 rows x 16 pairs
            int p = tid + i * 256;
            int r = p >> 4, kp = p & 15;
            if (A_BF16) {
                aR[i] = Ab[((size_t)(rowBase + r) * K + k0) / 2 + kp];
            } else {
                const float* s = Af + (size_t)(rowBase + r) * K + k0 + kp * 2;
                aR[i] = pack2(s[0], s[1]);
            }
        }
#pragma unroll
        for (int i = 0; i < 8; ++i) {                 // W: 128 rows x 16 pairs
            int p = tid + i * 256;
            int r = p >> 4, kp = p & 15;
            const float* s = W + (size_t)(colBase + r) * K + k0 + kp * 2;
            wR[i] = pack2(s[0], s[1]);
        }
    };
    auto store_tiles = [&](int buf) {
#pragma unroll
        for (int i = 0; i < 8; ++i) {
            int p = tid + i * 256;
            int r = p >> 4, kp = p & 15;
            As[buf][r][kp] = aR[i];
            Ws[buf][r][kp] = wR[i];
        }
    };

    load_tiles(0);
    store_tiles(0);
    __syncthreads();

    int buf = 0;
    for (int k0 = 0; k0 < K; k0 += BK) {
        const bool more = (k0 + BK) < K;
        if (more) load_tiles(k0 + BK);                // global -> regs (overlap)

        // A fragments: two 16-row slices of this wave's 32 rows
        FragU a0, a1;
        const int am = waveM * 32 + l16;
#pragma unroll
        for (int v = 0; v < 8; ++v) {
            int kp = (v & 3) + half * 4 + (v >> 2) * 8;
            a0.u[v] = As[buf][am][kp];
            a1.u[v] = As[buf][am + 16][kp];
        }
#pragma unroll
        for (int j = 0; j < 4; ++j) {
            FragU bf;
            const int bn = waveN * 64 + j * 16 + l16;
#pragma unroll
            for (int v = 0; v < 8; ++v) {
                int kp = (v & 3) + half * 4 + (v >> 2) * 8;
                bf.u[v] = Ws[buf][bn][kp];
            }
            acc[0][j] = __builtin_amdgcn_wmma_f32_16x16x32_bf16(
                false, a0.b, false, bf.b, (short)0, acc[0][j], false, false);
            acc[1][j] = __builtin_amdgcn_wmma_f32_16x16x32_bf16(
                false, a1.b, false, bf.b, (short)0, acc[1][j], false, false);
        }

        if (more) store_tiles(buf ^ 1);               // regs -> other LDS buffer
        __syncthreads();
        buf ^= 1;
    }

    // epilogue: bias + store.  C-layout: row m = v + half*8, col = lane%16
#pragma unroll
    for (int j = 0; j < 4; ++j) {
        const int n  = colBase + waveN * 64 + j * 16 + l16;
        const float bv = bias[n];
#pragma unroll
        for (int mi = 0; mi < 2; ++mi)
#pragma unroll
            for (int v = 0; v < 8; ++v) {
                const int m = rowBase + waveM * 32 + mi * 16 + v + half * 8;
                const float val = acc[mi][j][v] + bv;
                if (SCATTER) {
                    int b = m / T, t = m % T;
                    int h = n / HD, dd = n % HD;
                    ((unsigned short*)Dptr)[(((size_t)(b * H + h) * T + t) * HD) + dd] = f2bf(val);
                } else {
                    ((float*)Dptr)[(size_t)m * N + n] = val;
                }
            }
    }
}

// ---------------------------------------------------------------------------
// Flash attention: per block = 128 query rows of one (b,h).
// Qh/Kh/Vh are bf16 in [b*H+h][t][64] layout. Out is bf16 [b][t][C].
// ---------------------------------------------------------------------------
__global__ __launch_bounds__(256)
void attn_wmma(const unsigned short* __restrict__ Qh,
               const unsigned short* __restrict__ Kh,
               const unsigned short* __restrict__ Vh,
               unsigned short* __restrict__ Out,
               int Tq, int S, int H, int C)
{
    constexpr int BQ = 128, BS = 64, HD = 64;
    constexpr float scale = 0.125f;  // 1/sqrt(64)
    constexpr int KT_N = BS * 33;    // [s][d-pair], padded stride 33 u32
    constexpr int VT_N = HD * 33;    // transposed [d][s-pair]

    // Flat shared array so LDS byte offsets are explicit (static LDS base = 0).
    __shared__ unsigned int smem[KT_N + VT_N + BQ * 33];
    unsigned int* Kt = smem;                 // LDS byte offset 0
    unsigned int* Vt = smem + KT_N;
    unsigned int* Pt = smem + KT_N + VT_N;
    unsigned short* VtU = (unsigned short*)Vt;
    unsigned short* PtU = (unsigned short*)Pt;

    const int tid  = threadIdx.x;
    const int wave = tid >> 5;
    const int lane = tid & 31;
    const int l16  = lane & 15;
    const int half = lane >> 4;
    const int bh    = blockIdx.y;
    const int qbase = blockIdx.x * BQ;

    const unsigned int* Q32 = (const unsigned int*)Qh;
    const unsigned int* V32 = (const unsigned int*)Vh;
    const unsigned long long Kbase = (unsigned long long)(const void*)Kh;

    // Q fragments (held in registers for the whole kernel): 16x64 per wave
    FragU qf[2];
    {
        const int qm = qbase + wave * 16 + l16;
        const size_t qrow = ((size_t)bh * Tq + qm) * (HD / 2);
#pragma unroll
        for (int kc = 0; kc < 2; ++kc)
#pragma unroll
            for (int v = 0; v < 8; ++v) {
                int kp = kc * 16 + (v & 3) + half * 4 + (v >> 2) * 8;
                qf[kc].u[v] = Q32[qrow + kp];
            }
    }

    float mrow[8], lrow[8];
    v8f o[4];
#pragma unroll
    for (int v = 0; v < 8; ++v) { mrow[v] = -1e30f; lrow[v] = 0.0f; }
#pragma unroll
    for (int j = 0; j < 4; ++j)
#pragma unroll
        for (int e = 0; e < 8; ++e) o[j][e] = 0.0f;

    for (int s0 = 0; s0 < S; s0 += BS) {
        // ---- stage K chunk via async global->LDS (ASYNCcnt) ----
#pragma unroll
        for (int i = 0; i < 8; ++i) {
            int p = tid + i * 256;
            int r = p >> 5, dp = p & 31;
            unsigned lds_off = (unsigned)((r * 33 + dp) * 4);
            unsigned gb = (unsigned)((((size_t)bh * S + s0 + r) * 32 + dp) * 4);
            async_load_b32(lds_off, Kbase, gb);
        }
        // ---- stage V chunk transposed (needs VALU shuffle of halves) ----
#pragma unroll
        for (int i = 0; i < 8; ++i) {
            int p = tid + i * 256;
            int r = p >> 5, dp = p & 31;
            unsigned int pv = V32[((size_t)bh * S + s0 + r) * 32 + dp];
            int d0 = dp * 2;
            VtU[(size_t)d0 * 66 + r]       = (unsigned short)(pv & 0xffffu);
            VtU[(size_t)(d0 + 1) * 66 + r] = (unsigned short)(pv >> 16);
        }
        wait_async();
        __syncthreads();

        // ---- S = scale * Q @ K^T : 4 tiles of 16x16, K-dim = 64 (2 steps) ----
        v8f sc[4];
#pragma unroll
        for (int j = 0; j < 4; ++j) {
            FragU bk0, bk1;
            const int bn = j * 16 + l16;
#pragma unroll
            for (int v = 0; v < 8; ++v) {
                int kp = (v & 3) + half * 4 + (v >> 2) * 8;
                bk0.u[v] = Kt[bn * 33 + kp];
                bk1.u[v] = Kt[bn * 33 + 16 + kp];
            }
            v8f z;
#pragma unroll
            for (int e = 0; e < 8; ++e) z[e] = 0.0f;
            z = __builtin_amdgcn_wmma_f32_16x16x32_bf16(false, qf[0].b, false, bk0.b, (short)0, z, false, false);
            z = __builtin_amdgcn_wmma_f32_16x16x32_bf16(false, qf[1].b, false, bk1.b, (short)0, z, false, false);
            sc[j] = z;
        }

        // ---- online softmax over the 64-key chunk ----
        float mnew[8], alpha[8];
#pragma unroll
        for (int v = 0; v < 8; ++v) {
#pragma unroll
            for (int j = 0; j < 4; ++j) sc[j][v] *= scale;
            float mv = fmaxf(fmaxf(sc[0][v], sc[1][v]), fmaxf(sc[2][v], sc[3][v]));
#pragma unroll
            for (int mask = 1; mask < 16; mask <<= 1)
                mv = fmaxf(mv, __shfl_xor(mv, mask, 32));
            mnew[v]  = fmaxf(mrow[v], mv);
            alpha[v] = __expf(mrow[v] - mnew[v]);
            mrow[v]  = mnew[v];
        }
#pragma unroll
        for (int v = 0; v < 8; ++v) {
            float rs = 0.0f;
#pragma unroll
            for (int j = 0; j < 4; ++j) {
                float p = __expf(sc[j][v] - mnew[v]);
                sc[j][v] = p;
                rs += p;
            }
#pragma unroll
            for (int mask = 1; mask < 16; mask <<= 1)
                rs += __shfl_xor(rs, mask, 32);
            lrow[v] = lrow[v] * alpha[v] + rs;
#pragma unroll
            for (int j = 0; j < 4; ++j) o[j][v] *= alpha[v];
        }

        // ---- P (C-layout) -> LDS as bf16 A-layout source ----
#pragma unroll
        for (int v = 0; v < 8; ++v) {
            const int pr = wave * 16 + v + half * 8;
            unsigned short* Pr = PtU + (size_t)pr * 66;
#pragma unroll
            for (int j = 0; j < 4; ++j) Pr[j * 16 + l16] = f2bf(sc[j][v]);
        }
        __syncthreads();

        // ---- O += P @ V ----
        FragU pf[2];
        {
            const int pm = wave * 16 + l16;
#pragma unroll
            for (int kc = 0; kc < 2; ++kc)
#pragma unroll
                for (int v = 0; v < 8; ++v) {
                    int kp = kc * 16 + (v & 3) + half * 4 + (v >> 2) * 8;
                    pf[kc].u[v] = Pt[pm * 33 + kp];
                }
        }
#pragma unroll
        for (int j = 0; j < 4; ++j) {
            FragU vf0, vf1;
            const int vd = j * 16 + l16;
#pragma unroll
            for (int v = 0; v < 8; ++v) {
                int kp = (v & 3) + half * 4 + (v >> 2) * 8;
                vf0.u[v] = Vt[vd * 33 + kp];
                vf1.u[v] = Vt[vd * 33 + 16 + kp];
            }
            o[j] = __builtin_amdgcn_wmma_f32_16x16x32_bf16(false, pf[0].b, false, vf0.b, (short)0, o[j], false, false);
            o[j] = __builtin_amdgcn_wmma_f32_16x16x32_bf16(false, pf[1].b, false, vf1.b, (short)0, o[j], false, false);
        }
        __syncthreads();
    }

    // ---- normalize + write bf16 into [b, t, C] for the output projection ----
    const int b = bh / H, h = bh % H;
#pragma unroll
    for (int v = 0; v < 8; ++v) {
        const float inv = 1.0f / lrow[v];
        const int t = qbase + wave * 16 + v + half * 8;
        const size_t rowoff = ((size_t)b * Tq + t) * C + (size_t)h * HD;
#pragma unroll
        for (int j = 0; j < 4; ++j)
            Out[rowoff + j * 16 + l16] = f2bf(o[j][v] * inv);
    }
}

// ---------------------------------------------------------------------------
// Host-side launch
// ---------------------------------------------------------------------------
extern "C" void kernel_launch(void* const* d_in, const int* in_sizes, int n_in,
                              void* d_out, int out_size, void* d_ws, size_t ws_size,
                              hipStream_t stream)
{
    (void)in_sizes; (void)n_in; (void)out_size; (void)ws_size;

    constexpr int B  = 2, T = 2048, C = 1024, H = 16, HD = 64;
    constexpr int M  = B * T;                 // 4096
    constexpr size_t HEADSZ = (size_t)B * H * T * HD;  // 4,194,304 bf16 elems

    const float* q  = (const float*)d_in[0];
    const float* k  = (const float*)d_in[1];
    const float* v  = (const float*)d_in[2];
    const float* Wq = (const float*)d_in[3];
    const float* bq = (const float*)d_in[4];
    const float* Wk = (const float*)d_in[5];
    const float* bk = (const float*)d_in[6];
    const float* Wv = (const float*)d_in[7];
    const float* bv = (const float*)d_in[8];
    const float* Wo = (const float*)d_in[9];
    const float* bo = (const float*)d_in[10];

    unsigned short* Qh = (unsigned short*)d_ws;
    unsigned short* Kh = Qh + HEADSZ;
    unsigned short* Vh = Kh + HEADSZ;
    unsigned short* AO = Vh + HEADSZ;   // attention output, bf16 [M, C]

    dim3 gGemm(M / 128, C / 128);       // (32, 8)
    dim3 gAttn(T / 128, B * H);         // (16, 32)
    dim3 blk(256);

    // Q/K/V projections: fp32 in -> bf16 head-layout out
    gemm_wmma<false, true><<<gGemm, blk, 0, stream>>>(q, Wq, bq, Qh, M, C, C, T, H, HD);
    gemm_wmma<false, true><<<gGemm, blk, 0, stream>>>(k, Wk, bk, Kh, M, C, C, T, H, HD);
    gemm_wmma<false, true><<<gGemm, blk, 0, stream>>>(v, Wv, bv, Vh, M, C, C, T, H, HD);

    // flash attention
    attn_wmma<<<gAttn, blk, 0, stream>>>(Qh, Kh, Vh, AO, T, T, H, C);

    // output projection: bf16 in -> fp32 out
    gemm_wmma<true, false><<<gGemm, blk, 0, stream>>>(AO, Wo, bo, (float*)d_out,
                                                      M, C, C, 1, 1, 1);
}
